// DualAttentionLayer_40450001994244
// MI455X (gfx1250) — matrix operand order repaired and checked by reading, not compiled, not run on hardware
//
#include <hip/hip_runtime.h>
#include <cstdint>

typedef __attribute__((ext_vector_type(16))) __bf16 v16bf;
typedef __attribute__((ext_vector_type(8)))  float  v8f;

#define BB 4
#define SS 8192
#define EE 256
#define HH 8
#define DD 32
#define RR 128
#define XX 512
#define GG 64
#define MM (BB*SS)
#define KVSPLIT 16

#define ACT_NONE  0
#define ACT_RELU6 1
#define ACT_ELU1  2

// ---- CDNA5 async global->LDS copy support (toolchains differ; guard everything)
#if defined(__has_builtin)
#if __has_builtin(__builtin_amdgcn_global_load_async_to_lds_b128)
#define HAVE_ASYNC_LDS 1
#endif
#endif
#ifndef HAVE_ASYNC_LDS
#define HAVE_ASYNC_LDS 0
#endif

#if HAVE_ASYNC_LDS
typedef int v4i_g __attribute__((vector_size(16)));            // matches builtin's pointee type
typedef __attribute__((address_space(1))) v4i_g* as1_v4i;      // global (prints as __device__)
typedef __attribute__((address_space(3))) v4i_g* as3_v4i;      // LDS
__device__ __forceinline__ void async_cp16(const __bf16* g, __bf16* l) {
  v4i_g* gg = (v4i_g*)g;   // reinterpret (generic), drop const
  v4i_g* ll = (v4i_g*)l;
  __builtin_amdgcn_global_load_async_to_lds_b128((as1_v4i)gg, (as3_v4i)ll, 0, 0);
}
#endif

__device__ __forceinline__ void wait_async0() {
#if defined(__has_builtin) && __has_builtin(__builtin_amdgcn_s_wait_asynccnt)
  __builtin_amdgcn_s_wait_asynccnt(0);
#else
  asm volatile("s_wait_asynccnt 0x0" ::: "memory");
#endif
}

// ---------------------------------------------------------------- utility
__global__ void cvt_kernel(const float* __restrict__ s, __bf16* __restrict__ d, int n) {
  int i = blockIdx.x * blockDim.x + threadIdx.x;
  int stride = gridDim.x * blockDim.x;
  for (; i < n; i += stride) d[i] = (__bf16)s[i];
}

__global__ void zero_kernel(float* __restrict__ p, int n) {
  int i = blockIdx.x * blockDim.x + threadIdx.x;
  if (i < n) p[i] = 0.f;
}

// ---------------------------------------------------------------- WMMA GEMM
// C[M,N] = act( A[M,K] @ W[N,K]^T + bias + (accf ? C : 0) )
// A, W bf16 row-major (lda/ldw in elements). outF fp32 and/or outB bf16.
// 128x128 block tile, 8 waves (2x4), 64x32 per wave, K-step 32.
// Double-buffered LDS fed by async global->LDS DMA (ASYNCcnt) when available.
union Frag { uint4 u[2]; v16bf v; };

__global__ __launch_bounds__(256) void gemm_wmma(
    const __bf16* __restrict__ A, int lda,
    const __bf16* __restrict__ W, int ldw,
    const float* __restrict__ bias,
    float* __restrict__ outF, __bf16* __restrict__ outB,
    int N, int K, int act, int accf)
{
  __shared__ __bf16 lA[2][128 * 40];  // 128 rows x 32 halves, stride 40 (16B-aligned rows)
  __shared__ __bf16 lB[2][128 * 40];

  const int tid  = threadIdx.x;
  const int lane = tid & 31;
  const int wave = tid >> 5;
  const int wm   = wave & 1;   // 2 waves in M
  const int wn   = wave >> 1;  // 4 waves in N
  const int m0   = blockIdx.x * 128;
  const int n0   = blockIdx.y * 128;

  // per-thread tile segment: 32 contiguous bytes of one row
  const int  lrow   = tid >> 1;
  const int  lk     = (tid & 1) * 16;
  const bool wvalid = (n0 + lrow) < N;
  const int  lofs   = lrow * 40 + lk;

  // rows of the W tile beyond N stay zero for the whole kernel
  if (!wvalid) {
    uint4 z = make_uint4(0u, 0u, 0u, 0u);
    *reinterpret_cast<uint4*>(&lB[0][lofs])     = z;
    *reinterpret_cast<uint4*>(&lB[0][lofs + 8]) = z;
    *reinterpret_cast<uint4*>(&lB[1][lofs])     = z;
    *reinterpret_cast<uint4*>(&lB[1][lofs + 8]) = z;
  }

  const __bf16* gA = A + (size_t)(m0 + lrow) * lda + lk;
  const __bf16* gW = W + (size_t)(n0 + lrow) * ldw + lk;

  v8f acc[4][2];
  #pragma unroll
  for (int mi = 0; mi < 4; ++mi)
    #pragma unroll
    for (int ni = 0; ni < 2; ++ni)
      #pragma unroll
      for (int vi = 0; vi < 8; ++vi) acc[mi][ni][vi] = 0.f;

  const int nk = K >> 5;

#if HAVE_ASYNC_LDS
  // prologue: DMA first K-tile into buffer 0
  async_cp16(gA,     &lA[0][lofs]);
  async_cp16(gA + 8, &lA[0][lofs + 8]);
  if (wvalid) {
    async_cp16(gW,     &lB[0][lofs]);
    async_cp16(gW + 8, &lB[0][lofs + 8]);
  }
#else
  uint4 ra0, ra1;
  uint4 rw0 = make_uint4(0u, 0u, 0u, 0u), rw1 = rw0;
  ra0 = *reinterpret_cast<const uint4*>(gA);
  ra1 = *reinterpret_cast<const uint4*>(gA + 8);
  if (wvalid) {
    rw0 = *reinterpret_cast<const uint4*>(gW);
    rw1 = *reinterpret_cast<const uint4*>(gW + 8);
  }
#endif

  for (int ki = 0; ki < nk; ++ki) {
    const int buf = ki & 1;

#if HAVE_ASYNC_LDS
    wait_async0();          // this wave's DMA into `buf` done
    __syncthreads();        // everyone's DMA done; prior compute from buf^1 done
    if (ki + 1 < nk) {      // overlap next tile's DMA with this tile's WMMAs
      const int k0 = (ki + 1) << 5;
      async_cp16(gA + k0,     &lA[buf ^ 1][lofs]);
      async_cp16(gA + k0 + 8, &lA[buf ^ 1][lofs + 8]);
      if (wvalid) {
        async_cp16(gW + k0,     &lB[buf ^ 1][lofs]);
        async_cp16(gW + k0 + 8, &lB[buf ^ 1][lofs + 8]);
      }
    }
#else
    __syncthreads();
    *reinterpret_cast<uint4*>(&lA[buf][lofs])     = ra0;
    *reinterpret_cast<uint4*>(&lA[buf][lofs + 8]) = ra1;
    if (wvalid) {
      *reinterpret_cast<uint4*>(&lB[buf][lofs])     = rw0;
      *reinterpret_cast<uint4*>(&lB[buf][lofs + 8]) = rw1;
    }
    __syncthreads();
    if (ki + 1 < nk) {      // overlap next tile's global loads with compute
      const int k0 = (ki + 1) << 5;
      ra0 = *reinterpret_cast<const uint4*>(gA + k0);
      ra1 = *reinterpret_cast<const uint4*>(gA + k0 + 8);
      if (wvalid) {
        rw0 = *reinterpret_cast<const uint4*>(gW + k0);
        rw1 = *reinterpret_cast<const uint4*>(gW + k0 + 8);
      }
    }
#endif

    // A frag: lane row=lane&15; lo-lanes K{0..7,16..23}, hi-lanes K{8..15,24..31}
    // B frag: lane col=lane&15; lo-lanes K{0..7,16..23}, hi-lanes K{8..15,24..31}
    Frag af[4], bfr[2];
    const int ak = (lane >> 4) * 8;
    const int bk = (lane >> 4) * 8;
    #pragma unroll
    for (int mi = 0; mi < 4; ++mi) {
      int r = wm * 64 + mi * 16 + (lane & 15);
      af[mi].u[0] = *reinterpret_cast<const uint4*>(&lA[buf][r * 40 + ak]);
      af[mi].u[1] = *reinterpret_cast<const uint4*>(&lA[buf][r * 40 + ak + 16]);
    }
    #pragma unroll
    for (int ni = 0; ni < 2; ++ni) {
      int r = wn * 32 + ni * 16 + (lane & 15);
      bfr[ni].u[0] = *reinterpret_cast<const uint4*>(&lB[buf][r * 40 + bk]);
      bfr[ni].u[1] = *reinterpret_cast<const uint4*>(&lB[buf][r * 40 + bk + 16]);
    }
    #pragma unroll
    for (int mi = 0; mi < 4; ++mi)
      #pragma unroll
      for (int ni = 0; ni < 2; ++ni)
        acc[mi][ni] = __builtin_amdgcn_wmma_f32_16x16x32_bf16(
            false, af[mi].v, false, bfr[ni].v, (short)0, acc[mi][ni], false, false);
  }

  // epilogue: C layout v[i] -> M = i + 8*(lane>>4), N = lane&15
  #pragma unroll
  for (int mi = 0; mi < 4; ++mi) {
    #pragma unroll
    for (int ni = 0; ni < 2; ++ni) {
      int gcol = n0 + wn * 32 + ni * 16 + (lane & 15);
      if (gcol >= N) continue;
      float bv = bias ? bias[gcol] : 0.f;
      #pragma unroll
      for (int vi = 0; vi < 8; ++vi) {
        int grow = m0 + wm * 64 + mi * 16 + (lane >> 4) * 8 + vi;
        size_t idx = (size_t)grow * N + gcol;
        float v = acc[mi][ni][vi] + bv;
        if (accf) v += outF[idx];
        if (act == ACT_RELU6)      v = fminf(fmaxf(v, 0.f), 6.f);
        else if (act == ACT_ELU1)  v = (v > 0.f) ? (v + 1.f) : __expf(v);
        if (outF) outF[idx] = v;
        if (outB) outB[idx] = (__bf16)v;
      }
    }
  }
}

// ---------------------------------------------------------------- attention core
// kv[b,h,d,e] = sum_n k[b,n,h,d]*v[b,n,h,e]; ksum[b,h,d] = sum_n k[b,n,h,d]
__global__ __launch_bounds__(256) void kv_reduce_kernel(
    const float* __restrict__ kact, const float* __restrict__ vact,
    float* __restrict__ kv, float* __restrict__ ksum)
{
  __shared__ float ks[32][33];
  __shared__ float vs[32][33];
  int tid = threadIdx.x;
  int bh  = blockIdx.x / KVSPLIT;
  int sp  = blockIdx.x % KVSPLIT;
  int b   = bh >> 3;
  int h   = bh & 7;
  int rows  = SS / KVSPLIT;
  int nbase = b * SS + sp * rows;
  int e = tid & 31, w = tid >> 5;
  float a0 = 0, a1 = 0, a2 = 0, a3 = 0, sacc = 0;
  for (int nn = 0; nn < rows; nn += 32) {
    __syncthreads();
    #pragma unroll
    for (int t = 0; t < 4; ++t) {
      int idx = tid + t * 256;
      int r = idx >> 5, c = idx & 31;
      size_t g = (size_t)(nbase + nn + r) * EE + h * DD + c;
      ks[r][c] = kact[g];
      vs[r][c] = vact[g];
    }
    __syncthreads();
    #pragma unroll 4
    for (int j = 0; j < 32; ++j) {
      float ve = vs[j][e];
      a0 += ks[j][w * 4 + 0] * ve;
      a1 += ks[j][w * 4 + 1] * ve;
      a2 += ks[j][w * 4 + 2] * ve;
      a3 += ks[j][w * 4 + 3] * ve;
    }
    if (tid < 32) {
      #pragma unroll 4
      for (int j = 0; j < 32; ++j) sacc += ks[j][tid];
    }
  }
  float* kvp = kv + (size_t)bh * DD * DD;
  atomicAdd(&kvp[(w * 4 + 0) * DD + e], a0);
  atomicAdd(&kvp[(w * 4 + 1) * DD + e], a1);
  atomicAdd(&kvp[(w * 4 + 2) * DD + e], a2);
  atomicAdd(&kvp[(w * 4 + 3) * DD + e], a3);
  if (tid < 32) atomicAdd(&ksum[bh * DD + tid], sacc);
}

// out[b,n,h,e] = (sum_d q*kv) / (sum_d q*ksum + 1e-6)
__global__ __launch_bounds__(256) void attn_core_kernel(
    const float* __restrict__ qact, const float* __restrict__ kv,
    const float* __restrict__ ksum, __bf16* __restrict__ outB)
{
  __shared__ float qs[EE];
  int token = blockIdx.x;
  int tid = threadIdx.x;
  qs[tid] = qact[(size_t)token * EE + tid];
  __syncthreads();
  int b = token / SS;
  int h = tid >> 5, e = tid & 31;
  const float* kvp = kv + (size_t)(b * HH + h) * DD * DD;
  const float* ksp = ksum + (b * HH + h) * DD;
  float num = 0.f, den = 0.f;
  #pragma unroll 8
  for (int d = 0; d < DD; ++d) {
    float qd = qs[h * DD + d];
    num += qd * kvp[d * DD + e];
    den += qd * ksp[d];
  }
  outB[(size_t)token * EE + tid] = (__bf16)(num / (den + 1e-6f));
}

// ---------------------------------------------------------------- fused LN
// mode 0: x = a + b ; mode 1: x = a*gate[t,0] + b*gate[t,1]
__global__ __launch_bounds__(256) void ln_kernel(
    const float* __restrict__ a, const float* __restrict__ bsrc,
    const float* __restrict__ gate, const float* __restrict__ gamma,
    const float* __restrict__ beta, float* __restrict__ outF,
    __bf16* __restrict__ outB, int mode)
{
  __shared__ float red[16];
  int t = blockIdx.x, i = threadIdx.x;
  size_t idx = (size_t)t * EE + i;
  float x;
  if (mode == 0) x = a[idx] + bsrc[idx];
  else {
    float g0 = gate[t * 2 + 0], g1 = gate[t * 2 + 1];
    x = a[idx] * g0 + bsrc[idx] * g1;
  }
  float s1 = x, s2 = x * x;
  #pragma unroll
  for (int off = 16; off > 0; off >>= 1) {
    s1 += __shfl_xor(s1, off, 32);
    s2 += __shfl_xor(s2, off, 32);
  }
  int w = i >> 5;
  if ((i & 31) == 0) { red[w] = s1; red[8 + w] = s2; }
  __syncthreads();
  float t1 = 0.f, t2 = 0.f;
  #pragma unroll
  for (int k = 0; k < 8; ++k) { t1 += red[k]; t2 += red[8 + k]; }
  float mean = t1 * (1.f / EE);
  float var  = t2 * (1.f / EE) - mean * mean;
  float y = (x - mean) * rsqrtf(var + 1e-5f) * gamma[i] + beta[i];
  if (outF) outF[idx] = y;
  if (outB) outB[idx] = (__bf16)y;
}

// ---------------------------------------------------------------- gate logits + softmax(2)
__global__ __launch_bounds__(256) void gate2_kernel(
    const __bf16* __restrict__ gh, const float* __restrict__ w2,
    const float* __restrict__ b2, float* __restrict__ gate)
{
  int t = blockIdx.x * blockDim.x + threadIdx.x;
  if (t >= MM) return;
  const __bf16* g = gh + (size_t)t * GG;
  float l0 = b2[0], l1 = b2[1];
  #pragma unroll 8
  for (int i = 0; i < GG; ++i) {
    float x = (float)g[i];
    l0 += x * w2[i];
    l1 += x * w2[GG + i];
  }
  float m = fmaxf(l0, l1);
  float e0 = __expf(l0 - m), e1 = __expf(l1 - m);
  float inv = 1.f / (e0 + e1);
  gate[t * 2 + 0] = e0 * inv;
  gate[t * 2 + 1] = e1 * inv;
}

// ---------------------------------------------------------------- depthwise conv(k=3) + BN + relu6
__global__ __launch_bounds__(512) void dwconv_kernel(
    const __bf16* __restrict__ h1, const float* __restrict__ cw,
    const float* __restrict__ cb, const float* __restrict__ bg,
    const float* __restrict__ bb2, const float* __restrict__ bm,
    const float* __restrict__ bv, __bf16* __restrict__ h2)
{
  int token = blockIdx.x;
  int c = threadIdx.x;
  int n = token % SS;
  size_t base = (size_t)token * XX + c;
  float xm = (n > 0)      ? (float)h1[base - XX] : 0.f;
  float xc = (float)h1[base];
  float xp = (n < SS - 1) ? (float)h1[base + XX] : 0.f;
  float y = xm * cw[c * 3 + 0] + xc * cw[c * 3 + 1] + xp * cw[c * 3 + 2] + cb[c];
  y = (y - bm[c]) * rsqrtf(bv[c] + 1e-5f) * bg[c] + bb2[c];
  y = fminf(fmaxf(y, 0.f), 6.f);
  h2[base] = (__bf16)y;
}

// ---------------------------------------------------------------- host
static void launch_gemm(const __bf16* A, int lda, const __bf16* W, int ldw,
                        const float* bias, float* outF, __bf16* outB,
                        int M, int N, int K, int act, int accf, hipStream_t s)
{
  dim3 grid(M / 128, (N + 127) / 128);
  gemm_wmma<<<grid, dim3(256), 0, s>>>(A, lda, W, ldw, bias, outF, outB, N, K, act, accf);
}

extern "C" void kernel_launch(void* const* d_in, const int* in_sizes, int n_in,
                              void* d_out, int out_size, void* d_ws, size_t ws_size,
                              hipStream_t stream)
{
  (void)in_sizes; (void)n_in; (void)out_size;
  // JAX pytree (sorted-key) flattening of setup_inputs():
  // 0 body_feats, 1 limb_feats, then params:
  // attn b2l@2, l2b@13, self_body@24, self_limb@35
  //   sub: 0 k_down,1 k_up_b,2 k_up_w,3 out_b,4 out_w,5 q_down,6 q_up_b,7 q_up_w,8 v_down,9 v_up_b,10 v_up_w
  // ffn_body@46, ffn_limb@56: 0 b1,1 b2,2 bn_b,3 bn_g,4 bn_m,5 bn_v,6 conv_b,7 conv_w,8 w1,9 w2
  // gate@66: 0 b1,1 b2,2 w1,3 w2 ; ln@70: n1_body{b,g} n1_limb{b,g} n2_body{b,g} n3_body{b,g} n3_limb{b,g}
  const int FFN_BODY = 46, FFN_LIMB = 56, GATE = 66, LN0 = 70;
  auto F = [&](int i) -> const float* { return (const float*)d_in[i]; };

  float* out_body = (float*)d_out;
  float* out_limb = out_body + (size_t)MM * EE;

  size_t off = 0;
  auto alloc = [&](size_t bytes) -> void* {
    void* p = (char*)d_ws + off;
    off += (bytes + 255) & ~(size_t)255;
    return p;
  };

  // bf16 weights
  struct AttnW { __bf16 *qd, *kd, *vd, *qu, *ku, *vu, *ow;
                 const float *qub, *kub, *vub, *ob; };
  const int abase[4] = {24, 35, 2, 13};  // self_body, self_limb, b2l, l2b
  AttnW aw[4];
  for (int a = 0; a < 4; ++a) {
    aw[a].qd = (__bf16*)alloc(RR * EE * 2);
    aw[a].kd = (__bf16*)alloc(RR * EE * 2);
    aw[a].vd = (__bf16*)alloc(RR * EE * 2);
    aw[a].qu = (__bf16*)alloc(EE * RR * 2);
    aw[a].ku = (__bf16*)alloc(EE * RR * 2);
    aw[a].vu = (__bf16*)alloc(EE * RR * 2);
    aw[a].ow = (__bf16*)alloc(EE * EE * 2);
    int b = abase[a];
    aw[a].kub = F(b + 1); aw[a].ob = F(b + 3); aw[a].qub = F(b + 6); aw[a].vub = F(b + 9);
  }
  __bf16 *fw1[2], *fw2[2];
  fw1[0] = (__bf16*)alloc(XX * EE * 2); fw2[0] = (__bf16*)alloc(EE * XX * 2);
  fw1[1] = (__bf16*)alloc(XX * EE * 2); fw2[1] = (__bf16*)alloc(EE * XX * 2);
  __bf16* gw1 = (__bf16*)alloc(GG * 2 * EE * 2);

  // activations / scratch
  __bf16* bodyB   = (__bf16*)alloc((size_t)MM * EE * 2);
  __bf16* limbB   = (__bf16*)alloc((size_t)MM * EE * 2);
  __bf16* qlow    = (__bf16*)alloc((size_t)MM * RR * 2);
  __bf16* klow    = (__bf16*)alloc((size_t)MM * RR * 2);
  __bf16* vlow    = (__bf16*)alloc((size_t)MM * RR * 2);
  float*  qact    = (float*) alloc((size_t)MM * EE * 4);
  float*  kact    = (float*) alloc((size_t)MM * EE * 4);
  float*  vact    = (float*) alloc((size_t)MM * EE * 4);
  const int KVN   = BB * HH * DD * DD + BB * HH * DD;
  float*  kvbuf   = (float*) alloc((size_t)KVN * 4);
  __bf16* attnO   = (__bf16*)alloc((size_t)MM * EE * 2);
  float*  selfO   = (float*) alloc((size_t)MM * EE * 4);
  float*  body1_f = (float*) alloc((size_t)MM * EE * 4);
  float*  limb1_f = (float*) alloc((size_t)MM * EE * 4);
  __bf16* body1_b = (__bf16*)alloc((size_t)MM * EE * 2);
  __bf16* limb1_b = (__bf16*)alloc((size_t)MM * EE * 2);
  float*  bcross_f= (float*) alloc((size_t)MM * EE * 4);
  float*  lcross_f= (float*) alloc((size_t)MM * EE * 4);
  float*  gh_f    = (float*) alloc((size_t)MM * GG * 4);
  __bf16* gh_b    = (__bf16*)alloc((size_t)MM * GG * 2);
  float*  gateB   = (float*) alloc((size_t)MM * 2 * 4);
  float*  gateL   = (float*) alloc((size_t)MM * 2 * 4);
  __bf16* body2_b = (__bf16*)alloc((size_t)MM * EE * 2);
  __bf16* limb2_b = (__bf16*)alloc((size_t)MM * EE * 2);
  // aliases (lifetime-safe reuse)
  __bf16* bcross_b = bodyB;            // bodyB dead after self-attn
  __bf16* lcross_b = limbB;
  float*  body2_f  = qact;             // q/k/vact dead after last attn_core
  float*  limb2_f  = kact;
  float*  ffn_f    = vact;
  __bf16* h1       = (__bf16*)selfO;   // selfO dead after n1 LNs (same byte size)
  __bf16* h2       = (__bf16*)body1_f; // body1_f dead after n2 combine

  if (off > ws_size) return;  // workspace too small: nothing safe to do

  auto cvtp = [&](const float* src, __bf16* dst, int n) {
    int blocks = (n + 1023) >> 10;
    cvt_kernel<<<blocks, 256, 0, stream>>>(src, dst, n);
  };

  // weight + input conversions to bf16
  cvtp(F(0), bodyB, MM * EE);
  cvtp(F(1), limbB, MM * EE);
  for (int a = 0; a < 4; ++a) {
    int b = abase[a];
    cvtp(F(b + 5),  aw[a].qd, RR * EE);
    cvtp(F(b + 0),  aw[a].kd, RR * EE);
    cvtp(F(b + 8),  aw[a].vd, RR * EE);
    cvtp(F(b + 7),  aw[a].qu, EE * RR);
    cvtp(F(b + 2),  aw[a].ku, EE * RR);
    cvtp(F(b + 10), aw[a].vu, EE * RR);
    cvtp(F(b + 4),  aw[a].ow, EE * EE);
  }
  cvtp(F(FFN_BODY + 8), fw1[0], XX * EE);
  cvtp(F(FFN_BODY + 9), fw2[0], EE * XX);
  cvtp(F(FFN_LIMB + 8), fw1[1], XX * EE);
  cvtp(F(FFN_LIMB + 9), fw2[1], EE * XX);
  cvtp(F(GATE + 2), gw1, GG * 2 * EE);

  auto run_attn = [&](const AttnW& w, const __bf16* qx, const __bf16* kx, const __bf16* vx,
                      float* oF, __bf16* oB) {
    launch_gemm(qx, EE, w.qd, EE, nullptr, nullptr, qlow, MM, RR, EE, ACT_NONE, 0, stream);
    launch_gemm(kx, EE, w.kd, EE, nullptr, nullptr, klow, MM, RR, EE, ACT_NONE, 0, stream);
    launch_gemm(vx, EE, w.vd, EE, nullptr, nullptr, vlow, MM, RR, EE, ACT_NONE, 0, stream);
    launch_gemm(qlow, RR, w.qu, RR, w.qub, qact, nullptr, MM, EE, RR, ACT_ELU1, 0, stream);
    launch_gemm(klow, RR, w.ku, RR, w.kub, kact, nullptr, MM, EE, RR, ACT_ELU1, 0, stream);
    launch_gemm(vlow, RR, w.vu, RR, w.vub, vact, nullptr, MM, EE, RR, ACT_NONE, 0, stream);
    zero_kernel<<<(KVN + 255) / 256, 256, 0, stream>>>(kvbuf, KVN);
    kv_reduce_kernel<<<BB * HH * KVSPLIT, 256, 0, stream>>>(kact, vact, kvbuf,
                                                            kvbuf + BB * HH * DD * DD);
    attn_core_kernel<<<MM, 256, 0, stream>>>(qact, kvbuf, kvbuf + BB * HH * DD * DD, attnO);
    launch_gemm(attnO, EE, w.ow, EE, w.ob, oF, oB, MM, EE, EE, ACT_NONE, 0, stream);
  };

  // self attention + n1 LN (both streams)
  run_attn(aw[0], bodyB, bodyB, bodyB, selfO, nullptr);
  ln_kernel<<<MM, 256, 0, stream>>>(F(0), selfO, nullptr, F(LN0 + 1), F(LN0 + 0),
                                    body1_f, body1_b, 0);
  run_attn(aw[1], limbB, limbB, limbB, selfO, nullptr);
  ln_kernel<<<MM, 256, 0, stream>>>(F(1), selfO, nullptr, F(LN0 + 3), F(LN0 + 2),
                                    limb1_f, limb1_b, 0);

  // cross attention
  run_attn(aw[2], body1_b, limb1_b, limb1_b, bcross_f, bcross_b);  // b2l
  run_attn(aw[3], limb1_b, body1_b, body1_b, lcross_f, lcross_b);  // l2b

  // fusion gates: h = relu6([x, xc] @ w1^T + b1) split as two strided GEMMs
  launch_gemm(body1_b, EE, gw1, 2 * EE, nullptr, gh_f, nullptr, MM, GG, EE, ACT_NONE, 0, stream);
  launch_gemm(bcross_b, EE, gw1 + EE, 2 * EE, F(GATE + 0), gh_f, gh_b, MM, GG, EE, ACT_RELU6, 1, stream);
  gate2_kernel<<<(MM + 255) / 256, 256, 0, stream>>>(gh_b, F(GATE + 3), F(GATE + 1), gateB);
  launch_gemm(limb1_b, EE, gw1, 2 * EE, nullptr, gh_f, nullptr, MM, GG, EE, ACT_NONE, 0, stream);
  launch_gemm(lcross_b, EE, gw1 + EE, 2 * EE, F(GATE + 0), gh_f, gh_b, MM, GG, EE, ACT_RELU6, 1, stream);
  gate2_kernel<<<(MM + 255) / 256, 256, 0, stream>>>(gh_b, F(GATE + 3), F(GATE + 1), gateL);

  // gated combine + n2 (reference uses n2_body for BOTH streams)
  ln_kernel<<<MM, 256, 0, stream>>>(body1_f, bcross_f, gateB, F(LN0 + 5), F(LN0 + 4),
                                    body2_f, body2_b, 1);
  ln_kernel<<<MM, 256, 0, stream>>>(limb1_f, lcross_f, gateL, F(LN0 + 5), F(LN0 + 4),
                                    limb2_f, limb2_b, 1);

  // FFN body + n3_body -> d_out
  launch_gemm(body2_b, EE, fw1[0], EE, F(FFN_BODY + 0), nullptr, h1, MM, XX, EE, ACT_RELU6, 0, stream);
  dwconv_kernel<<<MM, 512, 0, stream>>>(h1, F(FFN_BODY + 7), F(FFN_BODY + 6), F(FFN_BODY + 3),
                                        F(FFN_BODY + 2), F(FFN_BODY + 4), F(FFN_BODY + 5), h2);
  launch_gemm(h2, XX, fw2[0], XX, F(FFN_BODY + 1), ffn_f, nullptr, MM, EE, XX, ACT_NONE, 0, stream);
  ln_kernel<<<MM, 256, 0, stream>>>(body2_f, ffn_f, nullptr, F(LN0 + 7), F(LN0 + 6),
                                    out_body, nullptr, 0);

  // FFN limb + n3_limb -> d_out
  launch_gemm(limb2_b, EE, fw1[1], EE, F(FFN_LIMB + 0), nullptr, h1, MM, XX, EE, ACT_RELU6, 0, stream);
  dwconv_kernel<<<MM, 512, 0, stream>>>(h1, F(FFN_LIMB + 7), F(FFN_LIMB + 6), F(FFN_LIMB + 3),
                                        F(FFN_LIMB + 2), F(FFN_LIMB + 4), F(FFN_LIMB + 5), h2);
  launch_gemm(h2, XX, fw2[1], XX, F(FFN_LIMB + 1), ffn_f, nullptr, MM, EE, XX, ACT_NONE, 0, stream);
  ln_kernel<<<MM, 256, 0, stream>>>(limb2_f, ffn_f, nullptr, F(LN0 + 9), F(LN0 + 8),
                                    out_limb, nullptr, 0);
}
